// _GRUD_2413771620922
// MI455X (gfx1250) — matrix operand order repaired and checked by reading, not compiled, not run on hardware
//
#include <hip/hip_runtime.h>
#include <hip/hip_bf16.h>

// GRU-D forward for MI455X (gfx1250, wave32, WMMA bf16).
// B=1024, T=128, F=128, H=256, D_IN=512.
// GEMM waves compute 32x64 macro-tiles (2x4 WMMA subtiles) for 8 WMMA per
// 12 b128 loads per k-step (register-level A/B fragment reuse).

#define Bsz 1024
#define Tsz 128
#define Fsz 128
#define Hsz 256
#define BT  (Bsz * Tsz)
#define DIN (2 * Fsz + Hsz)   // 512

typedef __attribute__((ext_vector_type(16))) __bf16          v16bf;
typedef __attribute__((ext_vector_type(8)))  float           v8f;
typedef __attribute__((ext_vector_type(8)))  unsigned short  ushort8;

__device__ __forceinline__ unsigned short f2bf(float f) {
    // round-to-nearest-even fp32 -> bf16 (bits)
    unsigned int u = __float_as_uint(f);
    u += 0x7FFFu + ((u >> 16) & 1u);
    return (unsigned short)(u >> 16);
}
__device__ __forceinline__ __bf16 bf_bits(unsigned short s) {
    return __builtin_bit_cast(__bf16, s);
}

// A fragment (16x32 bf16, ISA 7.12.2): lane m=L&15; element i<8 -> k=koff+i,
// element i>=8 -> k=koff+16+(i-8), koff=8*(L>=16). Two 16B loads.
__device__ __forceinline__ v16bf load_fragA(const unsigned short* p) {
    ushort8 c0 = *reinterpret_cast<const ushort8*>(p);
    ushort8 c1 = *reinterpret_cast<const ushort8*>(p + 16);
    v16bf a;
#pragma unroll
    for (int i = 0; i < 8; ++i) {
        a[i]     = bf_bits(c0[i]);
        a[i + 8] = bf_bits(c1[i]);
    }
    return a;
}
// B fragment (32x16 bf16): lane n=L&15; element i<8 -> k=koff+i,
// element i>=8 -> k=koff+16+(i-8), koff=8*(L>=16). Two 16B loads.
__device__ __forceinline__ v16bf load_fragB(const unsigned short* p) {
    ushort8 c0 = *reinterpret_cast<const ushort8*>(p);
    ushort8 c1 = *reinterpret_cast<const ushort8*>(p + 16);
    v16bf b;
#pragma unroll
    for (int i = 0; i < 8; ++i) {
        b[i]     = bf_bits(c0[i]);
        b[i + 8] = bf_bits(c1[i]);
    }
    return b;
}

// ---------------------------------------------------------------------------
// Weight repack: fp32 -> bf16, with column reordering so the time loop only
// sees one recurrent GEMM.
//   wgh  [H,F]      = W_gamma_h
//   wxm  [3H,2F]    = W_ih columns [x | m]          (non-recurrent part)
//   wrec [6H,H]     = stack(W_ih columns [h], W_hh) (recurrent part)
//   wo   [F,H]      = W_out
// ---------------------------------------------------------------------------
__global__ void grud_pack_weights(const float* __restrict__ Wgh,
                                  const float* __restrict__ Wih,
                                  const float* __restrict__ Whh,
                                  const float* __restrict__ Wout,
                                  unsigned short* __restrict__ wgh,
                                  unsigned short* __restrict__ wxm,
                                  unsigned short* __restrict__ wrec,
                                  unsigned short* __restrict__ wo) {
    int i = blockIdx.x * blockDim.x + threadIdx.x;
    const int n0 = Hsz * Fsz;          // 32768
    const int n1 = 3 * Hsz * 2 * Fsz;  // 196608
    const int n2 = 6 * Hsz * Hsz;      // 393216
    const int n3 = Fsz * Hsz;          // 32768
    if (i < n0) { wgh[i] = f2bf(Wgh[i]); return; }
    i -= n0;
    if (i < n1) {
        int r = i / (2 * Fsz), c = i % (2 * Fsz);
        int col = (c < Fsz) ? c : (Fsz + Hsz + (c - Fsz));
        wxm[i] = f2bf(Wih[r * DIN + col]);
        return;
    }
    i -= n1;
    if (i < n2) {
        int r = i / Hsz, c = i % Hsz;
        float v = (r < 3 * Hsz) ? Wih[r * DIN + Fsz + c]
                                : Whh[(r - 3 * Hsz) * Hsz + c];
        wrec[i] = f2bf(v);
        return;
    }
    i -= n2;
    if (i < n3) wo[i] = f2bf(Wout[i]);
}

// ---------------------------------------------------------------------------
// Batched elementwise prep over all B*T*F elements:
//   gamma_x = exp(-relu(d*diag(Wgx) + bgx))
//   x_h     = gamma_x*x_locf + (1-gamma_x)*emp_mean
//   x_rep   = m*x + (1-m)*x_h
// Emits bf16 rows [x_rep | m] (K=2F for the input GEMM) and bf16 deltas.
// ---------------------------------------------------------------------------
__global__ void grud_prep(const float* __restrict__ X,
                          const float* __restrict__ Mm,
                          const float* __restrict__ D,
                          const float* __restrict__ emp,
                          const float* __restrict__ Xl,
                          const float* __restrict__ Wgx,
                          const float* __restrict__ bgx,
                          unsigned short* __restrict__ xm_bf,
                          unsigned short* __restrict__ d_bf) {
    long idx = (long)blockIdx.x * blockDim.x + threadIdx.x;
    if (idx >= (long)BT * Fsz) return;
    int  f   = (int)(idx % Fsz);
    long row = idx / Fsz;
    int  b   = (int)(row / Tsz);
    float d  = D[idx];
    float gx = __expf(-fmaxf(d * Wgx[f * Fsz + f] + bgx[f], 0.0f));
    float xh = gx * Xl[idx] + (1.0f - gx) * emp[b * Fsz + f];
    float m  = Mm[idx];
    float xr = m * X[idx] + (1.0f - m) * xh;
    xm_bf[row * (2 * Fsz) + f]       = f2bf(xr);
    xm_bf[row * (2 * Fsz) + Fsz + f] = f2bf(m);
    d_bf[idx] = f2bf(d);
}

// ---------------------------------------------------------------------------
// Macro-tiled WMMA GEMM: C[M,N] = act( A[M,K](bf16) @ W[N,K]^T(bf16) + bias )
// block = 128 threads = 4 waves; block covers 128 M x 64 N; each wave owns a
// 32x64 macro-tile = 2(M) x 4(N) WMMA subtiles -> 8 wmma per 12 b128 loads.
// C/D layout: lane n=L&15; vgpr v -> m = v + 8*(L>=16).
// act: 0 = identity, 1 = exp(-relu(x))   (for gamma_h)
// Requires M % 128 == 0, N % 64 == 0, K % 32 == 0 (all shapes here comply).
// ---------------------------------------------------------------------------
__global__ __launch_bounds__(128)
void grud_wmma_gemm(const unsigned short* __restrict__ A, long lda,
                    const unsigned short* __restrict__ W, int K,
                    const float* __restrict__ bias,
                    float* __restrict__ C, long ldc,
                    int N, int act) {
    int lane = threadIdx.x & 31;
    int wave = threadIdx.x >> 5;
    int half = lane >> 4;
    int l15  = lane & 15;
    long m0 = (long)blockIdx.y * 128 + wave * 32;
    int  n0 = blockIdx.x * 64;

    const unsigned short* arow0 = A + (m0 + l15) * lda + half * 8;
    const unsigned short* arow1 = A + (m0 + 16 + l15) * lda + half * 8;
    const unsigned short* brow0 = W + (long)(n0 + l15) * K + half * 8;
    const unsigned short* brow1 = brow0 + 16L * K;
    const unsigned short* brow2 = brow0 + 32L * K;
    const unsigned short* brow3 = brow0 + 48L * K;

    v8f acc[2][4] = {};
    for (int k0 = 0; k0 < K; k0 += 32) {
        v16bf af0 = load_fragA(arow0 + k0);
        v16bf af1 = load_fragA(arow1 + k0);
        v16bf bf0 = load_fragB(brow0 + k0);
        v16bf bf1 = load_fragB(brow1 + k0);
        v16bf bf2 = load_fragB(brow2 + k0);
        v16bf bf3 = load_fragB(brow3 + k0);
        acc[0][0] = __builtin_amdgcn_wmma_f32_16x16x32_bf16(false, af0, false, bf0, (short)0, acc[0][0], false, false);
        acc[0][1] = __builtin_amdgcn_wmma_f32_16x16x32_bf16(false, af0, false, bf1, (short)0, acc[0][1], false, false);
        acc[0][2] = __builtin_amdgcn_wmma_f32_16x16x32_bf16(false, af0, false, bf2, (short)0, acc[0][2], false, false);
        acc[0][3] = __builtin_amdgcn_wmma_f32_16x16x32_bf16(false, af0, false, bf3, (short)0, acc[0][3], false, false);
        acc[1][0] = __builtin_amdgcn_wmma_f32_16x16x32_bf16(false, af1, false, bf0, (short)0, acc[1][0], false, false);
        acc[1][1] = __builtin_amdgcn_wmma_f32_16x16x32_bf16(false, af1, false, bf1, (short)0, acc[1][1], false, false);
        acc[1][2] = __builtin_amdgcn_wmma_f32_16x16x32_bf16(false, af1, false, bf2, (short)0, acc[1][2], false, false);
        acc[1][3] = __builtin_amdgcn_wmma_f32_16x16x32_bf16(false, af1, false, bf3, (short)0, acc[1][3], false, false);
    }

#pragma unroll
    for (int ni = 0; ni < 4; ++ni) {
        int   col = n0 + ni * 16 + l15;
        float bv  = bias ? bias[col] : 0.0f;
#pragma unroll
        for (int mi = 0; mi < 2; ++mi) {
            long mbase = m0 + mi * 16 + half * 8;
#pragma unroll
            for (int v = 0; v < 8; ++v) {
                float val = acc[mi][ni][v] + bv;
                if (act == 1) val = __expf(-fmaxf(val, 0.0f));
                C[(mbase + v) * ldc + col] = val;
            }
        }
    }
}

// ---------------------------------------------------------------------------
// Per-step: decay hidden state, record it (decayed pre-update h is what the
// reference collects) both as fp32 (for gate update) and bf16 (GEMM A / proj).
// ---------------------------------------------------------------------------
__global__ void grud_decay(float* __restrict__ h,
                           const float* __restrict__ gammah,
                           unsigned short* __restrict__ hid_bf, int t) {
    int i = blockIdx.x * blockDim.x + threadIdx.x;
    if (i >= Bsz * Hsz) return;
    int  b = i / Hsz, j = i % Hsz;
    long g = ((long)(b * Tsz + t)) * Hsz + j;
    float hv = h[i] * gammah[g];
    h[i] = hv;
    hid_bf[g] = f2bf(hv);
}

// ---------------------------------------------------------------------------
// Per-step GRU gate math.
//   GIX[row, 0:3H]   = [x_rep|m] @ W_ih_xm^T + b_ih     (precomputed)
//   GH [b, 0:3H]     = h_dec @ W_ih_h^T
//   GH [b, 3H:6H]    = h_dec @ W_hh^T                   (b_hh added here)
// ---------------------------------------------------------------------------
__global__ void grud_gate(const float* __restrict__ GIX,
                          const float* __restrict__ GH,
                          const float* __restrict__ bhh,
                          float* __restrict__ h, int t) {
    int i = blockIdx.x * blockDim.x + threadIdx.x;
    if (i >= Bsz * Hsz) return;
    int  b = i / Hsz, j = i % Hsz;
    long row = (long)b * Tsz + t;
    const float* gi = GIX + row * (3 * Hsz);
    const float* gh = GH + (long)b * (6 * Hsz);
    float ir  = gi[j]           + gh[j];
    float iz  = gi[Hsz + j]     + gh[Hsz + j];
    float in_ = gi[2 * Hsz + j] + gh[2 * Hsz + j];
    float hr  = gh[3 * Hsz + j] + bhh[j];
    float hz  = gh[4 * Hsz + j] + bhh[Hsz + j];
    float hn  = gh[5 * Hsz + j] + bhh[2 * Hsz + j];
    float r = 1.0f / (1.0f + __expf(-(ir + hr)));
    float z = 1.0f / (1.0f + __expf(-(iz + hz)));
    float n = tanhf(in_ + r * hn);
    h[i] = (1.0f - z) * n + z * h[i];
}

// ---------------------------------------------------------------------------
// Output projection WMMA GEMM (same 32x64 macro-tiling) with fused imputation
// and masked-MSE reduction.
//   recon   = hidden @ W_out^T + b_out
//   imputed = m*X + (1-m)*recon           -> d_out
//   accum[0] += sum(m*(recon-X)^2), accum[1] += sum(m)
// ---------------------------------------------------------------------------
__global__ __launch_bounds__(128)
void grud_proj(const unsigned short* __restrict__ Abf,
               const unsigned short* __restrict__ Wo,
               const float* __restrict__ bout,
               const float* __restrict__ X,
               const float* __restrict__ Mm,
               float* __restrict__ out,
               float* __restrict__ accum) {
    int lane = threadIdx.x & 31;
    int wave = threadIdx.x >> 5;
    int half = lane >> 4;
    int l15  = lane & 15;
    long m0 = (long)blockIdx.y * 128 + wave * 32;
    int  n0 = blockIdx.x * 64;

    const unsigned short* arow0 = Abf + (m0 + l15) * (long)Hsz + half * 8;
    const unsigned short* arow1 = Abf + (m0 + 16 + l15) * (long)Hsz + half * 8;
    const unsigned short* brow0 = Wo + (long)(n0 + l15) * Hsz + half * 8;
    const unsigned short* brow1 = brow0 + 16L * Hsz;
    const unsigned short* brow2 = brow0 + 32L * Hsz;
    const unsigned short* brow3 = brow0 + 48L * Hsz;

    v8f acc[2][4] = {};
    for (int k0 = 0; k0 < Hsz; k0 += 32) {
        v16bf af0 = load_fragA(arow0 + k0);
        v16bf af1 = load_fragA(arow1 + k0);
        v16bf bf0 = load_fragB(brow0 + k0);
        v16bf bf1 = load_fragB(brow1 + k0);
        v16bf bf2 = load_fragB(brow2 + k0);
        v16bf bf3 = load_fragB(brow3 + k0);
        acc[0][0] = __builtin_amdgcn_wmma_f32_16x16x32_bf16(false, af0, false, bf0, (short)0, acc[0][0], false, false);
        acc[0][1] = __builtin_amdgcn_wmma_f32_16x16x32_bf16(false, af0, false, bf1, (short)0, acc[0][1], false, false);
        acc[0][2] = __builtin_amdgcn_wmma_f32_16x16x32_bf16(false, af0, false, bf2, (short)0, acc[0][2], false, false);
        acc[0][3] = __builtin_amdgcn_wmma_f32_16x16x32_bf16(false, af0, false, bf3, (short)0, acc[0][3], false, false);
        acc[1][0] = __builtin_amdgcn_wmma_f32_16x16x32_bf16(false, af1, false, bf0, (short)0, acc[1][0], false, false);
        acc[1][1] = __builtin_amdgcn_wmma_f32_16x16x32_bf16(false, af1, false, bf1, (short)0, acc[1][1], false, false);
        acc[1][2] = __builtin_amdgcn_wmma_f32_16x16x32_bf16(false, af1, false, bf2, (short)0, acc[1][2], false, false);
        acc[1][3] = __builtin_amdgcn_wmma_f32_16x16x32_bf16(false, af1, false, bf3, (short)0, acc[1][3], false, false);
    }

    float lnum = 0.0f, lden = 0.0f;
#pragma unroll
    for (int ni = 0; ni < 4; ++ni) {
        int   col = n0 + ni * 16 + l15;
        float bv  = bout[col];
#pragma unroll
        for (int mi = 0; mi < 2; ++mi) {
            long mbase = m0 + mi * 16 + half * 8;
#pragma unroll
            for (int v = 0; v < 8; ++v) {
                float recon = acc[mi][ni][v] + bv;
                long  idx   = (mbase + v) * Fsz + col;
                float xv = X[idx], mv = Mm[idx];
                out[idx] = mv * xv + (1.0f - mv) * recon;
                float dd = recon - xv;
                lnum += mv * dd * dd;
                lden += mv;
            }
        }
    }
    // wave32 butterfly reduction, then one atomic pair per wave
    for (int off = 16; off > 0; off >>= 1) {
        lnum += __shfl_xor(lnum, off, 32);
        lden += __shfl_xor(lden, off, 32);
    }
    if (lane == 0) {
        atomicAdd(&accum[0], lnum);
        atomicAdd(&accum[1], lden);
    }
}

__global__ void grud_final(const float* __restrict__ accum, float* __restrict__ out) {
    if (threadIdx.x == 0 && blockIdx.x == 0)
        out[(long)BT * Fsz] = accum[0] / (accum[1] + 1e-12f);
}

// ---------------------------------------------------------------------------
extern "C" void kernel_launch(void* const* d_in, const int* in_sizes, int n_in,
                              void* d_out, int out_size, void* d_ws, size_t ws_size,
                              hipStream_t stream) {
    const float* X    = (const float*)d_in[0];
    const float* Mm   = (const float*)d_in[1];
    const float* D    = (const float*)d_in[2];
    const float* emp  = (const float*)d_in[3];
    const float* Xl   = (const float*)d_in[4];
    const float* Wgh  = (const float*)d_in[5];
    const float* bgh  = (const float*)d_in[6];
    const float* Wgx  = (const float*)d_in[7];
    const float* bgx  = (const float*)d_in[8];
    const float* Wih  = (const float*)d_in[9];
    const float* bih  = (const float*)d_in[10];
    const float* Whh  = (const float*)d_in[11];
    const float* bhh  = (const float*)d_in[12];
    const float* Wout = (const float*)d_in[13];
    const float* bout = (const float*)d_in[14];
    float* out = (float*)d_out;

    // workspace bump allocator (256B aligned)
    char* w = (char*)d_ws;
    auto alloc = [&](size_t bytes) -> void* {
        char* p = w;
        w += (bytes + 255) & ~(size_t)255;
        return p;
    };
    unsigned short* xm_bf  = (unsigned short*)alloc((size_t)BT * 2 * Fsz * 2);   //  64 MB
    unsigned short* d_bf   = (unsigned short*)alloc((size_t)BT * Fsz * 2);       //  32 MB
    float*          gammah = (float*)alloc((size_t)BT * Hsz * 4);                // 128 MB
    float*          GIX    = (float*)alloc((size_t)BT * 3 * Hsz * 4);            // 384 MB
    unsigned short* hid_bf = (unsigned short*)alloc((size_t)BT * Hsz * 2);       //  64 MB
    float*          hbuf   = (float*)alloc((size_t)Bsz * Hsz * 4);               //   1 MB
    float*          GH     = (float*)alloc((size_t)Bsz * 6 * Hsz * 4);           //   6 MB
    unsigned short* wgh    = (unsigned short*)alloc((size_t)Hsz * Fsz * 2);
    unsigned short* wxm    = (unsigned short*)alloc((size_t)3 * Hsz * 2 * Fsz * 2);
    unsigned short* wrec   = (unsigned short*)alloc((size_t)6 * Hsz * Hsz * 2);
    unsigned short* wo     = (unsigned short*)alloc((size_t)Fsz * Hsz * 2);
    float*          accum  = (float*)alloc(2 * sizeof(float));

    hipMemsetAsync(hbuf, 0, (size_t)Bsz * Hsz * 4, stream);  // h0 = 0
    hipMemsetAsync(accum, 0, 2 * sizeof(float), stream);     // loss accumulators

    // Phase A: weight repack + batched elementwise prep
    grud_pack_weights<<<(655360 + 255) / 256, 256, 0, stream>>>(
        Wgh, Wih, Whh, Wout, wgh, wxm, wrec, wo);
    grud_prep<<<(int)(((long)BT * Fsz + 255) / 256), 256, 0, stream>>>(
        X, Mm, D, emp, Xl, Wgx, bgx, xm_bf, d_bf);

    // Phase B: big non-recurrent GEMMs over all B*T rows
    // GIX = [x_rep|m] @ W_xm^T + b_ih            [131072 x 768], K=256
    grud_wmma_gemm<<<dim3(768 / 64, BT / 128), 128, 0, stream>>>(
        xm_bf, 2 * Fsz, wxm, 2 * Fsz, bih, GIX, 3 * Hsz, 3 * Hsz, 0);
    // gamma_h = exp(-relu(d @ Wgh^T + bgh))      [131072 x 256], K=128
    grud_wmma_gemm<<<dim3(256 / 64, BT / 128), 128, 0, stream>>>(
        d_bf, Fsz, wgh, Fsz, bgh, gammah, Hsz, Hsz, 1);

    // Phase C: sequential recurrence (critical path)
    for (int t = 0; t < Tsz; ++t) {
        grud_decay<<<(Bsz * Hsz + 255) / 256, 256, 0, stream>>>(
            hbuf, gammah, hid_bf, t);
        // GH = h_dec @ W_rec^T                   [1024 x 1536], K=256
        grud_wmma_gemm<<<dim3(1536 / 64, Bsz / 128), 128, 0, stream>>>(
            hid_bf + (size_t)t * Hsz, (long)Tsz * Hsz, wrec, Hsz,
            nullptr, GH, 6 * Hsz, 6 * Hsz, 0);
        grud_gate<<<(Bsz * Hsz + 255) / 256, 256, 0, stream>>>(
            GIX, GH, bhh, hbuf, t);
    }

    // Phase D: output projection + fused imputation & loss
    grud_proj<<<dim3(128 / 64, BT / 128), 128, 0, stream>>>(
        hid_bf, wo, bout, X, Mm, out, accum);
    grud_final<<<1, 32, 0, stream>>>(accum, out);
}
